// ScannedRNN_7541962571846
// MI455X (gfx1250) — matrix operand order repaired and checked by reading, not compiled, not run on hardware
//
#include <hip/hip_runtime.h>
#include <math.h>

// ---------------- problem dims ----------------
#define T_DIM 512
#define B_DIM 256
#define H_DIM 512
#define G3    1536              // 3*H
#define KT    (H_DIM / 32)      // 16 k-chunks of 32
#define NT    (G3 / 16)         // 96 column tiles of 16
#define WP_ELEMS (KT * NT * 32 * 16)   // packed bf16 elems per weight matrix

typedef __attribute__((ext_vector_type(16))) __bf16 v16bf;
typedef __attribute__((ext_vector_type(8)))  float  v8f;

union BF16x16 { v16bf v; unsigned short u[16]; uint4 q[2]; };

// accurate RNE fp32->bf16 (used in one-time pre-passes only)
__device__ __forceinline__ unsigned short f2bf_rne(float f) {
    unsigned int x = __builtin_bit_cast(unsigned int, f);
    x += 0x7fffu + ((x >> 16) & 1u);
    return (unsigned short)(x >> 16);
}
// fast hw-convert fp32->bf16 (hot path)
__device__ __forceinline__ unsigned short f2bf_hw(float f) {
    return __builtin_bit_cast(unsigned short, (__bf16)f);
}
__device__ __forceinline__ float bf2f(unsigned short h) {
    unsigned int x = ((unsigned int)h) << 16;
    return __builtin_bit_cast(float, x);
}
__device__ __forceinline__ float sigmoid_fast(float x) {
    return 1.0f / (1.0f + __expf(-x));
}
__device__ __forceinline__ float tanh_fast(float x) {
    float a = fabsf(x);
    float e = __expf(-2.0f * a);
    float t = (1.0f - e) / (1.0f + e);
    return copysignf(t, x);
}

__device__ __forceinline__ v8f wmma_bf16(v16bf a, v16bf b, v8f c) {
    return __builtin_amdgcn_wmma_f32_16x16x32_bf16(
        false, a, false, b, (short)0, c, false, false);
}

__device__ __forceinline__ v16bf load_bfrag(const unsigned short* __restrict__ p) {
    BF16x16 r;
    r.q[0] = ((const uint4*)p)[0];
    r.q[1] = ((const uint4*)p)[1];
    return r.v;
}

// -------- one-time weight repack: fp32 [H,3H] row-major -> WMMA B-fragment order --------
__global__ void pack_weights(const float* __restrict__ W, unsigned short* __restrict__ Wp) {
    int i = blockIdx.x * blockDim.x + threadIdx.x;
    if (i >= WP_ELEMS) return;
    int j    = i & 15;
    int lane = (i >> 4) & 31;
    int nt   = (i >> 9) % NT;
    int kt   = i / (16 * 32 * NT);
    int col  = nt * 16 + (lane & 15);
    int k    = kt * 32 + ((lane >= 16) ? 8 : 0) + (j & 7) + ((j >> 3) << 4);
    Wp[i] = f2bf_rne(W[(size_t)k * G3 + col]);
}

__global__ void zero_u16(unsigned short* __restrict__ p, int n) {
    int i = blockIdx.x * blockDim.x + threadIdx.x;
    if (i < n) p[i] = 0;
}

// -------- one-time bulk fp32 -> bf16 convert of X (8 elems / thread) --------
__global__ void cvt_x_bf16(const float* __restrict__ src, unsigned short* __restrict__ dst,
                           size_t n8) {
    size_t i = (size_t)blockIdx.x * blockDim.x + threadIdx.x;
    if (i >= n8) return;
    const float4* s = (const float4*)src + i * 2;
    float4 a = s[0], b = s[1];
    union { unsigned short u[8]; uint4 q; } r;
    r.u[0] = f2bf_rne(a.x); r.u[1] = f2bf_rne(a.y);
    r.u[2] = f2bf_rne(a.z); r.u[3] = f2bf_rne(a.w);
    r.u[4] = f2bf_rne(b.x); r.u[5] = f2bf_rne(b.y);
    r.u[6] = f2bf_rne(b.z); r.u[7] = f2bf_rne(b.w);
    ((uint4*)dst)[i] = r.q;
}

// -------- one GRU timestep --------
// grid = (H/16 = 32 column strips), block = 256 (8 waves). Each wave: 32 rows (two
// 16-row M-tiles sharing the six B fragments), 16 cols, all three gates + gating.
// h_in is ALREADY masked by resets[t] (stored pre-masked by the previous step),
// so the hot loop has zero masking work. h_out is stored masked by resets[t+1].
template <bool XPRE>
__global__ __launch_bounds__(256)
void gru_step(const float* __restrict__ x_t,               // [B,H] fp32 (fallback)
              const unsigned short* __restrict__ xb_t,     // [B,H] bf16 (XPRE)
              const unsigned char* __restrict__ resets_next, // [B] or nullptr
              const unsigned short* __restrict__ Wpi,
              const unsigned short* __restrict__ Wph,
              const float* __restrict__ bi,                // [3H]
              const float* __restrict__ bhn,               // [H]
              const unsigned short* __restrict__ h_in,     // bf16 [B,H], pre-masked
              unsigned short* __restrict__ h_out,          // bf16 [B,H]
              float* __restrict__ y_t)                     // [B,H] fp32
{
    const int lane  = threadIdx.x & 31;
    const int wave  = threadIdx.x >> 5;
    const int m0    = wave * 32;                 // rows m0 .. m0+31
    const int c0    = blockIdx.x * 16;
    const int ntile = blockIdx.x;
    const int colL  = lane & 15;
    const int hiK   = lane >> 4;                 // K-half select for A frags

    const int arow0 = m0 + colL;
    const int arow1 = m0 + 16 + colL;

    v8f accr0 = {}, accz0 = {}, accxn0 = {}, acchn0 = {};
    v8f accr1 = {}, accz1 = {}, accxn1 = {}, acchn1 = {};

    for (int kt = 0; kt < KT; ++kt) {
        const int k0 = kt * 32 + hiK * 8;

        // ---- A fragments from x ----
        BF16x16 ax0, ax1;
        if (XPRE) {
            const unsigned short* p0 = xb_t + (size_t)arow0 * H_DIM + k0;
            ax0.q[0] = *(const uint4*)p0;
            ax0.q[1] = *(const uint4*)(p0 + 16);
            const unsigned short* p1 = xb_t + (size_t)arow1 * H_DIM + k0;
            ax1.q[0] = *(const uint4*)p1;
            ax1.q[1] = *(const uint4*)(p1 + 16);
        } else {
            const float* f0 = x_t + (size_t)arow0 * H_DIM + k0;
            const float* f1 = x_t + (size_t)arow1 * H_DIM + k0;
            float4 a0 = ((const float4*)f0)[0], a1 = ((const float4*)f0)[1];
            float4 a2 = ((const float4*)(f0 + 16))[0], a3 = ((const float4*)(f0 + 16))[1];
            float4 b0 = ((const float4*)f1)[0], b1 = ((const float4*)f1)[1];
            float4 b2 = ((const float4*)(f1 + 16))[0], b3 = ((const float4*)(f1 + 16))[1];
            ax0.u[0]=f2bf_hw(a0.x); ax0.u[1]=f2bf_hw(a0.y); ax0.u[2]=f2bf_hw(a0.z); ax0.u[3]=f2bf_hw(a0.w);
            ax0.u[4]=f2bf_hw(a1.x); ax0.u[5]=f2bf_hw(a1.y); ax0.u[6]=f2bf_hw(a1.z); ax0.u[7]=f2bf_hw(a1.w);
            ax0.u[8]=f2bf_hw(a2.x); ax0.u[9]=f2bf_hw(a2.y); ax0.u[10]=f2bf_hw(a2.z); ax0.u[11]=f2bf_hw(a2.w);
            ax0.u[12]=f2bf_hw(a3.x); ax0.u[13]=f2bf_hw(a3.y); ax0.u[14]=f2bf_hw(a3.z); ax0.u[15]=f2bf_hw(a3.w);
            ax1.u[0]=f2bf_hw(b0.x); ax1.u[1]=f2bf_hw(b0.y); ax1.u[2]=f2bf_hw(b0.z); ax1.u[3]=f2bf_hw(b0.w);
            ax1.u[4]=f2bf_hw(b1.x); ax1.u[5]=f2bf_hw(b1.y); ax1.u[6]=f2bf_hw(b1.z); ax1.u[7]=f2bf_hw(b1.w);
            ax1.u[8]=f2bf_hw(b2.x); ax1.u[9]=f2bf_hw(b2.y); ax1.u[10]=f2bf_hw(b2.z); ax1.u[11]=f2bf_hw(b2.w);
            ax1.u[12]=f2bf_hw(b3.x); ax1.u[13]=f2bf_hw(b3.y); ax1.u[14]=f2bf_hw(b3.z); ax1.u[15]=f2bf_hw(b3.w);
        }

        // ---- A fragments from pre-masked h_{t-1} (bf16) ----
        BF16x16 ah0, ah1;
        {
            const unsigned short* p0 = h_in + (size_t)arow0 * H_DIM + k0;
            ah0.q[0] = *(const uint4*)p0;
            ah0.q[1] = *(const uint4*)(p0 + 16);
            const unsigned short* p1 = h_in + (size_t)arow1 * H_DIM + k0;
            ah1.q[0] = *(const uint4*)p1;
            ah1.q[1] = *(const uint4*)(p1 + 16);
        }

        // ---- B fragments (pre-packed, L2-resident) ----
        const size_t base = (size_t)kt * (NT * 32 * 16);
        const size_t lo   = (size_t)lane * 16;
        v16bf b_ir = load_bfrag(Wpi + base + (size_t)(ntile     ) * 512 + lo);
        v16bf b_iz = load_bfrag(Wpi + base + (size_t)(ntile + 32) * 512 + lo);
        v16bf b_in = load_bfrag(Wpi + base + (size_t)(ntile + 64) * 512 + lo);
        v16bf b_hr = load_bfrag(Wph + base + (size_t)(ntile     ) * 512 + lo);
        v16bf b_hz = load_bfrag(Wph + base + (size_t)(ntile + 32) * 512 + lo);
        v16bf b_hn = load_bfrag(Wph + base + (size_t)(ntile + 64) * 512 + lo);

        accr0  = wmma_bf16(ax0.v, b_ir, accr0);
        accr1  = wmma_bf16(ax1.v, b_ir, accr1);
        accz0  = wmma_bf16(ax0.v, b_iz, accz0);
        accz1  = wmma_bf16(ax1.v, b_iz, accz1);
        accxn0 = wmma_bf16(ax0.v, b_in, accxn0);
        accxn1 = wmma_bf16(ax1.v, b_in, accxn1);
        accr0  = wmma_bf16(ah0.v, b_hr, accr0);
        accr1  = wmma_bf16(ah1.v, b_hr, accr1);
        accz0  = wmma_bf16(ah0.v, b_hz, accz0);
        accz1  = wmma_bf16(ah1.v, b_hz, accz1);
        acchn0 = wmma_bf16(ah0.v, b_hn, acchn0);
        acchn1 = wmma_bf16(ah1.v, b_hn, acchn1);
    }

    // ---- gating ----
    const float bir  = bi[c0 + colL];
    const float biz  = bi[H_DIM + c0 + colL];
    const float bin  = bi[2 * H_DIM + c0 + colL];
    const float bhnv = bhn[c0 + colL];

    #pragma unroll
    for (int tile = 0; tile < 2; ++tile) {
        const v8f& ar = tile ? accr1 : accr0;
        const v8f& az = tile ? accz1 : accz0;
        const v8f& an = tile ? accxn1 : accxn0;
        const v8f& ah = tile ? acchn1 : acchn0;
        const int mt  = m0 + tile * 16;
        #pragma unroll
        for (int v = 0; v < 8; ++v) {
            const int    row = mt + v + hiK * 8;        // C/D: M = v + 8*(lane>=16)
            const size_t idx = (size_t)row * H_DIM + c0 + colL;
            const float hprev = bf2f(h_in[idx]);        // already reset-masked
            const float r  = sigmoid_fast(ar[v] + bir);
            const float z  = sigmoid_fast(az[v] + biz);
            const float n  = tanh_fast(an[v] + bin + r * (ah[v] + bhnv));
            const float hn = (1.0f - z) * n + z * hprev;
            y_t[idx] = hn;                               // unmasked output
            const bool rstn = resets_next && (resets_next[row] != 0);
            h_out[idx] = rstn ? (unsigned short)0 : f2bf_hw(hn);  // pre-mask for t+1
        }
    }
}

extern "C" void kernel_launch(void* const* d_in, const int* in_sizes, int n_in,
                              void* d_out, int out_size, void* d_ws, size_t ws_size,
                              hipStream_t stream) {
    (void)in_sizes; (void)n_in; (void)out_size;

    const float*         ins    = (const float*)d_in[0];          // [T,B,H]
    const unsigned char* resets = (const unsigned char*)d_in[1];  // [T,B] bool
    const float*         Wi     = (const float*)d_in[2];          // [H,3H]
    const float*         bi     = (const float*)d_in[3];          // [3H]
    const float*         Wh     = (const float*)d_in[4];          // [H,3H]
    const float*         bhn    = (const float*)d_in[5];          // [H]
    float*               out    = (float*)d_out;                  // [T,B,H]

    // ws layout (u16 elems): Wpi | Wph | hbuf0 | hbuf1 | [optional bf16 X]
    unsigned short* Wpi   = (unsigned short*)d_ws;
    unsigned short* Wph   = Wpi + WP_ELEMS;
    unsigned short* hbuf0 = Wph + WP_ELEMS;
    unsigned short* hbuf1 = hbuf0 + (size_t)B_DIM * H_DIM;
    unsigned short* xb    = hbuf1 + (size_t)B_DIM * H_DIM;

    const size_t base_elems = 2 * (size_t)WP_ELEMS + 2 * (size_t)B_DIM * H_DIM;
    const size_t x_elems    = (size_t)T_DIM * B_DIM * H_DIM;
    const bool   xpre       = ws_size >= (base_elems + x_elems) * sizeof(unsigned short);

    pack_weights<<<(WP_ELEMS + 255) / 256, 256, 0, stream>>>(Wi, Wpi);
    pack_weights<<<(WP_ELEMS + 255) / 256, 256, 0, stream>>>(Wh, Wph);
    zero_u16<<<(B_DIM * H_DIM + 255) / 256, 256, 0, stream>>>(hbuf0, B_DIM * H_DIM);
    if (xpre) {
        const size_t n8 = x_elems / 8;
        cvt_x_bf16<<<(unsigned)((n8 + 255) / 256), 256, 0, stream>>>(ins, xb, n8);
    }

    dim3 grid(H_DIM / 16);   // 32 column strips; 8 waves/block cover all 256 rows
    for (int t = 0; t < T_DIM; ++t) {
        const unsigned short* hin  = (t & 1) ? hbuf1 : hbuf0;
        unsigned short*       hout = (t & 1) ? hbuf0 : hbuf1;
        const unsigned char*  rstn = (t + 1 < T_DIM) ? resets + (size_t)(t + 1) * B_DIM
                                                     : (const unsigned char*)nullptr;
        if (xpre) {
            gru_step<true><<<grid, 256, 0, stream>>>(
                nullptr, xb + (size_t)t * B_DIM * H_DIM, rstn,
                Wpi, Wph, bi, bhn, hin, hout,
                out + (size_t)t * B_DIM * H_DIM);
        } else {
            gru_step<false><<<grid, 256, 0, stream>>>(
                ins + (size_t)t * B_DIM * H_DIM, nullptr, rstn,
                Wpi, Wph, bi, bhn, hin, hout,
                out + (size_t)t * B_DIM * H_DIM);
        }
    }
}
